// CEDLTwoLoop100M_34806414967460
// MI455X (gfx1250) — matrix operand order repaired and checked by reading, not compiled, not run on hardware
//
#include <hip/hip_runtime.h>
#include <hip/hip_bf16.h>
#include <stddef.h>

// ---------------- problem constants ----------------
#define BATCH   2
#define TSEQ    2048
#define DMODEL  640
#define NH      10
#define DH      64
#define MROWS   (BATCH * TSEQ)      // 4096
#define GN_EPS  1e-5f

// ---------------- WMMA types ----------------
typedef __attribute__((ext_vector_type(16))) __bf16 v16bf;
typedef __attribute__((ext_vector_type(8)))  float  v8f;

union BF16Frag {
    v16bf        v;
    unsigned int u[8];
};

__device__ __forceinline__ unsigned short f2bf(float f) {
    unsigned int u = __float_as_uint(f);
    unsigned int r = (u + 0x7FFFu + ((u >> 16) & 1u)) >> 16;   // RNE
    return (unsigned short)r;
}

__device__ __forceinline__ v8f wmma_bf16(const BF16Frag& a, const BF16Frag& b, v8f c) {
    return __builtin_amdgcn_wmma_f32_16x16x32_bf16(
        /*neg_a=*/false, a.v, /*neg_b=*/false, b.v,
        /*c_mod=*/(short)0, c, /*reuse_a=*/false, /*reuse_b=*/false);
}

// A fragment: 16x32 bf16 tile of row-major [row, k] array (ISA 7.12.2 layout).
// lane = h*16 + r : row = r ; v0..3 -> K = (h?8:0)+2v ; v4..7 -> K = 16+(h?8:0)+2(v-4)
// (two contiguous 16B runs per lane -> compiler merges into 2x global_load_b128)
__device__ __forceinline__ void load_frag_a(BF16Frag& f, const unsigned short* src,
                                            int ld, int row0, int k0) {
    const int lane = threadIdx.x & 31;
    const int h = lane >> 4, r = lane & 15;
    const unsigned short* p = src + (size_t)(row0 + r) * ld + k0;
#pragma unroll
    for (int v = 0; v < 8; ++v) {
        int kk = ((v < 4) ? 0 : 16) + (h ? 8 : 0) + 2 * (v & 3);
        f.u[v] = *(const unsigned int*)(p + kk);
    }
}

// B fragment: 32x16 (KxN) taken from row-major [n, k] array (i.e. B^T storage).
// lane = h*16 + n : col = n ; v0..3 -> K = (h?8:0)+2v ; v4..7 -> K = 16+(h?8:0)+2(v-4)
__device__ __forceinline__ void load_frag_b(BF16Frag& f, const unsigned short* src,
                                            int ld, int n0, int k0) {
    const int lane = threadIdx.x & 31;
    const int h = lane >> 4, n = lane & 15;
    const unsigned short* p = src + (size_t)(n0 + n) * ld + k0 + (h ? 8 : 0);
#pragma unroll
    for (int v = 0; v < 8; ++v)
        f.u[v] = *(const unsigned int*)(p + ((v < 4) ? 0 : 16) + 2 * (v & 3));
}

// ---------------- fp32 -> bf16 convert ----------------
__global__ void cvt_bf16_kernel(const float* __restrict__ in,
                                unsigned short* __restrict__ out, int n) {
    int i = blockIdx.x * blockDim.x + threadIdx.x;
    if (i < n) out[i] = f2bf(in[i]);
}

// ---------------- generic WMMA GEMM: C = A(MxK) * Bw(NxK)^T + bias ----------------
// 32x32 C macro-tile per wave: 2 A-frags x 2 B-frags -> 4 WMMAs per 32-deep k-step
// (1:1 fragment-load : wmma ratio, double the reuse of the 16x16 version).
// mode 0: fp32 store (row-major MxN)      -> outF
// mode 1: fp32 SiLU store                 -> outF
// mode 2: bf16 store, head-split (B,NH,T,DH)             -> outH
// mode 3: bf16 store, head-split transposed (B,NH,DH,T)  -> outH
__global__ void __launch_bounds__(256)
wmma_gemm_kernel(const unsigned short* __restrict__ A,
                 const unsigned short* __restrict__ Bw,
                 const float* __restrict__ bias,
                 float* __restrict__ outF, unsigned short* __restrict__ outH,
                 int M, int N, int Kc, int mode) {
    const int wave   = threadIdx.x >> 5;
    const int tile   = blockIdx.x * 8 + wave;
    const int tilesN = N >> 5;                  // 32-wide macro tiles
    const int ntiles = (M >> 5) * tilesN;
    if (tile >= ntiles) return;                 // wave-uniform: EXEC stays all-ones
    const int tm = tile / tilesN, tn = tile % tilesN;

    const v8f vzero = {};
    v8f acc[2][2];
#pragma unroll
    for (int mi = 0; mi < 2; ++mi)
#pragma unroll
        for (int ni = 0; ni < 2; ++ni) acc[mi][ni] = vzero;

    for (int kc = 0; kc < Kc; kc += 32) {
        BF16Frag a0, a1, b0, b1;
        load_frag_a(a0, A,  Kc, tm * 32,      kc);
        load_frag_a(a1, A,  Kc, tm * 32 + 16, kc);
        load_frag_b(b0, Bw, Kc, tn * 32,      kc);
        load_frag_b(b1, Bw, Kc, tn * 32 + 16, kc);
        acc[0][0] = wmma_bf16(a0, b0, acc[0][0]);
        acc[0][1] = wmma_bf16(a0, b1, acc[0][1]);
        acc[1][0] = wmma_bf16(a1, b0, acc[1][0]);
        acc[1][1] = wmma_bf16(a1, b1, acc[1][1]);
    }

    const int lane = threadIdx.x & 31;
    const int h = lane >> 4, n = lane & 15;
#pragma unroll
    for (int ni = 0; ni < 2; ++ni) {
        const int col = tn * 32 + ni * 16 + n;
        const float bs = bias[col];
#pragma unroll
        for (int mi = 0; mi < 2; ++mi) {
#pragma unroll
            for (int v = 0; v < 8; ++v) {
                const int row = tm * 32 + mi * 16 + v + 8 * h;
                float val = acc[mi][ni][v] + bs;
                if (mode == 0) {
                    outF[(size_t)row * N + col] = val;
                } else if (mode == 1) {
                    outF[(size_t)row * N + col] = val / (1.0f + __expf(-val)); // SiLU
                } else {
                    const int bidx = row >> 11;         // row / 2048
                    const int t    = row & 2047;
                    const int hd   = col >> 6;          // col / 64
                    const int ch   = col & 63;
                    if (mode == 2)
                        outH[(((size_t)bidx * NH + hd) * TSEQ + t) * DH + ch] = f2bf(val);
                    else
                        outH[(((size_t)bidx * NH + hd) * DH + ch) * TSEQ + t] = f2bf(val);
                }
            }
        }
    }
}

// ---------------- flash-style retention ----------------
// one wave per 32-row query tile of one (b,head); causal sweep over key blocks of 32.
// K/V fragments are shared across both 16-row query sub-tiles (16 WMMAs per 8
// fragment loads per key block). rowsum normalization deferred (linearity):
// O = (sum_j P_j V_j) / rowAbsSum.
__global__ void __launch_bounds__(32)
retention_kernel(const unsigned short* __restrict__ Qb,   // (B,NH,T,DH) bf16
                 const unsigned short* __restrict__ Kb,   // (B,NH,T,DH) bf16
                 const unsigned short* __restrict__ Vt,   // (B,NH,DH,T) bf16
                 float* __restrict__ Obuf,                // (B,NH,T,DH) f32
                 const float* __restrict__ gamma_log,
                 const float* __restrict__ log_lambda,
                 const float* __restrict__ phi) {
    const int bk   = blockIdx.y;
    const int head = bk % NH;
    const int i0   = blockIdx.x * 32;

    const float g     = 1.0f / (1.0f + __expf(-gamma_log[head]));
    const float lg    = __logf(g);
    const float wfreq = 6.28318530717958647f * __expf(-log_lambda[head]);
    const float ph    = phi[head];

    const unsigned short* q    = Qb + (size_t)bk * TSEQ * DH;
    const unsigned short* kmat = Kb + (size_t)bk * TSEQ * DH;
    const unsigned short* vt   = Vt + (size_t)bk * DH * TSEQ;

    const int lane = threadIdx.x & 31;
    const int h = lane >> 4, nn = lane & 15;

    BF16Frag aq[2][2];                         // Q tile resident: 2 sub-tiles x K-dim 64
    load_frag_a(aq[0][0], q, DH, i0,      0);
    load_frag_a(aq[0][1], q, DH, i0,      32);
    load_frag_a(aq[1][0], q, DH, i0 + 16, 0);
    load_frag_a(aq[1][1], q, DH, i0 + 16, 32);

    const v8f vzero = {};
    v8f o[2][4];
    float rowabs[2][8];
#pragma unroll
    for (int mt = 0; mt < 2; ++mt) {
#pragma unroll
        for (int nt = 0; nt < 4; ++nt) o[mt][nt] = vzero;
#pragma unroll
        for (int v = 0; v < 8; ++v) rowabs[mt][v] = 0.0f;
    }

    __shared__ __attribute__((aligned(4))) unsigned short pl[32 * 32];  // P repack

    for (int j0 = 0; j0 < i0 + 32; j0 += 32) {
        // K fragments for this key block (shared by both query sub-tiles)
        BF16Frag bk00, bk01, bk10, bk11;
        load_frag_b(bk00, kmat, DH, j0,      0);
        load_frag_b(bk01, kmat, DH, j0,      32);
        load_frag_b(bk10, kmat, DH, j0 + 16, 0);
        load_frag_b(bk11, kmat, DH, j0 + 16, 32);

#pragma unroll
        for (int mt = 0; mt < 2; ++mt) {
            // S = Q K^T over two 16-col sub-tiles (K-dim 64 = 2 wmma each)
            v8f s0 = vzero, s1 = vzero;
            s0 = wmma_bf16(aq[mt][0], bk00, s0);
            s0 = wmma_bf16(aq[mt][1], bk01, s0);
            s1 = wmma_bf16(aq[mt][0], bk10, s1);
            s1 = wmma_bf16(aq[mt][1], bk11, s1);

            // fused decay * periodic * causal mask; |row sum|; bf16 pack to LDS
#pragma unroll
            for (int v = 0; v < 8; ++v) {
                const int m  = mt * 16 + v + 8 * h;   // row within 32-row tile
                const int i  = i0 + m;
                const int d0 = i - (j0 + nn);
                const int d1 = d0 - 16;
                float f0 = (d0 < 0) ? 0.0f
                         : __expf(lg * (float)d0) * __cosf(wfreq * (float)d0 + ph);
                float f1 = (d1 < 0) ? 0.0f
                         : __expf(lg * (float)d1) * __cosf(wfreq * (float)d1 + ph);
                float p0 = s0[v] * f0, p1 = s1[v] * f1;
                rowabs[mt][v] += fabsf(p0) + fabsf(p1);
                pl[m * 32 + nn]      = f2bf(p0);
                pl[m * 32 + 16 + nn] = f2bf(p1);
            }
        }
        __syncthreads();

        // A-fragments of P (2 x 16x32) from LDS per ISA layout
        BF16Frag ap0, ap1;
        {
            const unsigned short* p0 = pl + (size_t)nn * 32;
            const unsigned short* p1 = pl + (size_t)(16 + nn) * 32;
#pragma unroll
            for (int v = 0; v < 8; ++v) {
                int kk = ((v < 4) ? 0 : 16) + (h ? 8 : 0) + 2 * (v & 3);
                ap0.u[v] = *(const unsigned int*)(p0 + kk);
                ap1.u[v] = *(const unsigned int*)(p1 + kk);
            }
        }

        // O += P * V  (V pre-transposed -> dword pair loads along key axis;
        // each V fragment feeds both query sub-tiles)
#pragma unroll
        for (int nt = 0; nt < 4; ++nt) {
            BF16Frag bv;
            load_frag_b(bv, vt, TSEQ, nt * 16, j0);
            o[0][nt] = wmma_bf16(ap0, bv, o[0][nt]);
            o[1][nt] = wmma_bf16(ap1, bv, o[1][nt]);
        }
        __syncthreads();
    }

    // reduce row |sum| across the 16 lanes of each half-wave (xor 1,2,4,8), scale, store
#pragma unroll
    for (int mt = 0; mt < 2; ++mt) {
        float inv[8];
#pragma unroll
        for (int v = 0; v < 8; ++v) {
            float rs = rowabs[mt][v];
            rs += __shfl_xor(rs, 1, 32);
            rs += __shfl_xor(rs, 2, 32);
            rs += __shfl_xor(rs, 4, 32);
            rs += __shfl_xor(rs, 8, 32);
            inv[v] = 1.0f / fmaxf(rs, 1.0f);
        }
        float* ob = Obuf + (size_t)bk * TSEQ * DH + (size_t)(i0 + mt * 16) * DH;
#pragma unroll
        for (int v = 0; v < 8; ++v) {
            const int m = v + 8 * h;
            ob[m * DH + nn]      = o[mt][0][v] * inv[v];
            ob[m * DH + 16 + nn] = o[mt][1][v] * inv[v];
            ob[m * DH + 32 + nn] = o[mt][2][v] * inv[v];
            ob[m * DH + 48 + nn] = o[mt][3][v] * inv[v];
        }
    }
}

// ---------------- GroupNorm (per b,head over DH*T) + SiLU-gate, bf16 repack ----------------
__global__ void __launch_bounds__(256)
groupnorm_gate_kernel(const float* __restrict__ Obuf,   // (B,NH,T,DH)
                      const float* __restrict__ gateS,  // (B,T,D) silu already applied
                      const float* __restrict__ gnw, const float* __restrict__ gnb,
                      unsigned short* __restrict__ Afinal) {  // (B*T, D) bf16
    const int bk = blockIdx.x;
    const int b  = bk / NH, hd = bk % NH;
    const int NEL = TSEQ * DH;                 // 131072
    const float* base = Obuf + (size_t)bk * NEL;
    const int tid = threadIdx.x;

    float s = 0.0f, ss = 0.0f;
    for (int i = tid; i < NEL; i += 256) {
        float x = base[i];
        s += x; ss += x * x;
    }
    __shared__ float r1[256], r2[256];
    r1[tid] = s; r2[tid] = ss;
    __syncthreads();
    for (int st = 128; st > 0; st >>= 1) {
        if (tid < st) { r1[tid] += r1[tid + st]; r2[tid] += r2[tid + st]; }
        __syncthreads();
    }
    const float mean = r1[0] / (float)NEL;
    const float var  = r2[0] / (float)NEL - mean * mean;
    const float rstd = rsqrtf(var + GN_EPS);

    for (int i = tid; i < NEL; i += 256) {
        const int t = i >> 6, ch = i & 63;
        const int col = hd * DH + ch;
        float val = (base[i] - mean) * rstd * gnw[col] + gnb[col];
        float gv  = gateS[((size_t)b * TSEQ + t) * DMODEL + col];
        Afinal[((size_t)b * TSEQ + t) * DMODEL + col] = f2bf(val * gv);
    }
}

// ---------------- host-side orchestration ----------------
extern "C" void kernel_launch(void* const* d_in, const int* in_sizes, int n_in,
                              void* d_out, int out_size, void* d_ws, size_t ws_size,
                              hipStream_t stream) {
    const float* x       = (const float*)d_in[0];
    const float* w_q_w   = (const float*)d_in[1];
    const float* w_q_b   = (const float*)d_in[2];
    const float* w_k_w   = (const float*)d_in[3];
    const float* w_k_b   = (const float*)d_in[4];
    const float* w_v_w   = (const float*)d_in[5];
    const float* w_v_b   = (const float*)d_in[6];
    const float* w_out_w = (const float*)d_in[7];
    const float* w_out_b = (const float*)d_in[8];
    const float* gate_w  = (const float*)d_in[9];
    const float* gate_b  = (const float*)d_in[10];
    const float* gn_w    = (const float*)d_in[11];
    const float* gn_b    = (const float*)d_in[12];
    const float* gamma_l = (const float*)d_in[13];
    const float* log_lam = (const float*)d_in[14];
    const float* phi     = (const float*)d_in[15];
    float* out = (float*)d_out;

    // workspace carve-up (all L2-resident: ~51 MB total vs 192 MB L2)
    char* base = (char*)d_ws;
    size_t off = 0;
    auto carve = [&](size_t bytes) -> void* {
        void* p = base + off;
        off += (bytes + 255) & ~(size_t)255;
        return p;
    };
    const size_t actN = (size_t)MROWS * DMODEL;        // 2,621,440
    const size_t wN   = (size_t)DMODEL * DMODEL;       //   409,600
    const size_t hN   = (size_t)BATCH * NH * TSEQ * DH;// 2,621,440

    unsigned short* xb  = (unsigned short*)carve(actN * 2);
    unsigned short* wqb = (unsigned short*)carve(wN * 2);
    unsigned short* wkb = (unsigned short*)carve(wN * 2);
    unsigned short* wvb = (unsigned short*)carve(wN * 2);
    unsigned short* wgb = (unsigned short*)carve(wN * 2);
    unsigned short* wob = (unsigned short*)carve(wN * 2);
    unsigned short* Qb  = (unsigned short*)carve(hN * 2);
    unsigned short* Kb  = (unsigned short*)carve(hN * 2);
    unsigned short* Vt  = (unsigned short*)carve(hN * 2);
    float*          gateS  = (float*)carve(actN * 4);
    float*          Obuf   = (float*)carve(hN * 4);
    unsigned short* Afinal = (unsigned short*)carve(actN * 2);
    (void)ws_size; (void)n_in; (void)in_sizes; (void)out_size;

    // 1) fp32 -> bf16 converts
    cvt_bf16_kernel<<<(int)((actN + 255) / 256), 256, 0, stream>>>(x, xb, (int)actN);
    cvt_bf16_kernel<<<(int)((wN + 255) / 256), 256, 0, stream>>>(w_q_w, wqb, (int)wN);
    cvt_bf16_kernel<<<(int)((wN + 255) / 256), 256, 0, stream>>>(w_k_w, wkb, (int)wN);
    cvt_bf16_kernel<<<(int)((wN + 255) / 256), 256, 0, stream>>>(w_v_w, wvb, (int)wN);
    cvt_bf16_kernel<<<(int)((wN + 255) / 256), 256, 0, stream>>>(gate_w, wgb, (int)wN);
    cvt_bf16_kernel<<<(int)((wN + 255) / 256), 256, 0, stream>>>(w_out_w, wob, (int)wN);

    // 2) projection GEMMs (bf16 WMMA, 32x32 macro-tiles, fused epilogues)
    const int tiles = (MROWS / 32) * (DMODEL / 32);    // 2560 macro-tiles
    const int gblk  = (tiles + 7) / 8;                 // 320 blocks of 8 waves
    wmma_gemm_kernel<<<gblk, 256, 0, stream>>>(xb, wqb, w_q_b, nullptr, Qb,
                                               MROWS, DMODEL, DMODEL, 2);
    wmma_gemm_kernel<<<gblk, 256, 0, stream>>>(xb, wkb, w_k_b, nullptr, Kb,
                                               MROWS, DMODEL, DMODEL, 2);
    wmma_gemm_kernel<<<gblk, 256, 0, stream>>>(xb, wvb, w_v_b, nullptr, Vt,
                                               MROWS, DMODEL, DMODEL, 3);
    wmma_gemm_kernel<<<gblk, 256, 0, stream>>>(xb, wgb, gate_b, gateS, nullptr,
                                               MROWS, DMODEL, DMODEL, 1);

    // 3) flash-style retention (never materializes the 335 MB score tensor)
    dim3 rgrid(TSEQ / 32, BATCH * NH);                 // (64, 20) -> 1280 waves
    retention_kernel<<<rgrid, 32, 0, stream>>>(Qb, Kb, Vt, Obuf,
                                               gamma_l, log_lam, phi);

    // 4) GroupNorm + SiLU gate fused, bf16 repack for final GEMM
    groupnorm_gate_kernel<<<BATCH * NH, 256, 0, stream>>>(Obuf, gateS, gn_w, gn_b, Afinal);

    // 5) output projection straight into d_out (fp32)
    wmma_gemm_kernel<<<gblk, 256, 0, stream>>>(Afinal, wob, w_out_b, out, nullptr,
                                               MROWS, DMODEL, DMODEL, 0);
}
